// Simple_Model_69604239999655
// MI455X (gfx1250) — compile-verified, matrix operand order checked by reading, not run on hardware
//
#include <hip/hip_runtime.h>

typedef __attribute__((ext_vector_type(16))) _Float16 v16h;
typedef __attribute__((ext_vector_type(8)))  float    v8f;

#define NH 512
#define NWAVES 16
#define TEMP_INV 100.0f

// ---------------------------------------------------------------------------
// Kernel 1: column sum of h [nRows, 512]  ->  partial[nBlocks, 512]
// Bandwidth-bound (204.8 MB @ 23.3 TB/s ~ 9us). 4-row unroll keeps 4
// independent loads in flight per lane; fixed partial layout + fixed-order
// final reduction => bit-deterministic (no float atomics).
// ---------------------------------------------------------------------------
__global__ __launch_bounds__(256)
void colsum_kernel(const float* __restrict__ h, float* __restrict__ partial,
                   int nRows, int rowsPerBlock)
{
    const int b = blockIdx.x, t = threadIdx.x;   // 256 threads, 2 cols each
    const int c = t * 2;
    int r0 = b * rowsPerBlock;
    int r1 = r0 + rowsPerBlock; if (r1 > nRows) r1 = nRows;
    float s0 = 0.f, s1 = 0.f;
    const float* p = h + (size_t)r0 * 512 + c;
    int r = r0;
    for (; r + 4 <= r1; r += 4, p += 2048) {
        __builtin_prefetch(p + 512 * 32, 0, 0);      // global_prefetch_b8
        float2 v0 = *(const float2*)(p);             // coalesced 2KB/row/block
        float2 v1 = *(const float2*)(p + 512);
        float2 v2 = *(const float2*)(p + 1024);
        float2 v3 = *(const float2*)(p + 1536);
        s0 += (v0.x + v1.x) + (v2.x + v3.x);
        s1 += (v0.y + v1.y) + (v2.y + v3.y);
    }
    for (; r < r1; ++r, p += 512) {
        float2 v = *(const float2*)p;
        s0 += v.x; s1 += v.y;
    }
    partial[(size_t)b * 512 + c]     = s0;
    partial[(size_t)b * 512 + c + 1] = s1;
}

// ---------------------------------------------------------------------------
// Wave-cooperative GEMV via V_WMMA_F32_16X16X32_F16.
//   y[row] (+)= sum_k W[row,k] * x[k]  + bias[row]
// A operand = x chunk broadcast across the 16 A-rows (per-lane K pattern),
// B operand = W^T tile (lane n holds output row tile*16+n, same K pattern).
// D rows are identical; outputs read from c[0] of lanes 0..15 (M=0, N=lane).
// Requirements: M % 16 == 0; x in LDS zero-padded to Kpad (multiple of 32);
// row stride ldw even (8B-aligned rows) for the vectorized fast path;
// called by ALL threads of the block (EXEC all-ones for WMMA).
// ---------------------------------------------------------------------------
__device__ inline void wmma_gemv(const float* __restrict__ W, int M, int ldw,
                                 int Kreal, int Kpad,
                                 const float* __restrict__ x,
                                 const float* __restrict__ bias,
                                 float* __restrict__ y,
                                 int wave, int lane, bool accumulate)
{
    const int n    = lane & 15;
    const int krow = (lane < 16) ? 0 : 8;
    const int nTiles = M >> 4;
    for (int tile = wave; tile < nTiles; tile += NWAVES) {
        const int row = tile * 16 + n;
        const float* __restrict__ wrow = W + (size_t)row * (size_t)ldw;
        v8f c = {0.f,0.f,0.f,0.f,0.f,0.f,0.f,0.f};
        for (int kb = 0; kb < Kpad; kb += 32) {
            v16h a, b;
            if (kb + 32 <= Kreal) {                    // fast path: vector loads
                const float*  wr = wrow + kb + krow;   // 8B-aligned (ldw, kb+krow even)
                const float2* xp = (const float2*)(x + kb + krow);
#pragma unroll
                for (int q = 0; q < 4; ++q) {
                    float2 x0 = xp[q];
                    float2 x1 = xp[8 + q];             // +16 floats
                    float2 w0 = *(const float2*)(wr + 2 * q);
                    float2 w1 = *(const float2*)(wr + 16 + 2 * q);
                    a[2*q]     = (_Float16)x0.x; a[2*q + 1]     = (_Float16)x0.y;
                    a[8 + 2*q] = (_Float16)x1.x; a[8 + 2*q + 1] = (_Float16)x1.y;
                    b[2*q]     = (_Float16)w0.x; b[2*q + 1]     = (_Float16)w0.y;
                    b[8 + 2*q] = (_Float16)w1.x; b[8 + 2*q + 1] = (_Float16)w1.y;
                }
            } else {                                   // tail: guarded scalar
#pragma unroll
                for (int j = 0; j < 16; ++j) {
                    const int k = kb + j + ((j >= 8) ? 8 : 0) + krow;
                    a[j] = (_Float16)x[k];
                    b[j] = (_Float16)((k < Kreal) ? wrow[k] : 0.0f);
                }
            }
            c = __builtin_amdgcn_wmma_f32_16x16x32_f16(
                    false, a, false, b, (short)0, c, false, false);
        }
        if (lane < 16) {
            float v = c[0] + (bias ? bias[row] : 0.0f);
            if (accumulate) y[row] += v; else y[row] = v;
        }
    }
}

__device__ inline float preluf(float x, float a) { return x >= 0.f ? x : a * x; }
__device__ inline float sigmoidf(float x) { return 1.0f / (1.0f + expf(-x)); }

// Deterministic hash-based gumbel noise (stand-in for JAX PRNG; same inputs
// -> same outputs on every call, which is the requirement here).
__device__ inline float gumbel_noise(unsigned idx, unsigned salt)
{
    unsigned x = idx * 0x9E3779B9u ^ (salt * 0x85EBCA6Bu + 0xC2B2AE35u);
    x ^= x >> 16; x *= 0x7FEB352Du; x ^= x >> 15; x *= 0x846CA68Bu; x ^= x >> 16;
    float u = (float)(x >> 8) * (1.0f / 16777216.0f);
    u = fmaxf(u, 1e-10f);
    return -logf(-logf(u));
}

__device__ inline void softmax_serial(float* v, int n)
{
    float m = v[0];
    for (int i = 1; i < n; ++i) m = fmaxf(m, v[i]);
    float s = 0.f;
    for (int i = 0; i < n; ++i) { v[i] = expf(v[i] - m); s += v[i]; }
    float inv = 1.0f / s;
    for (int i = 0; i < n; ++i) v[i] *= inv;
}

struct HeadParams {
    const float *goalVec, *goalObjectsVec, *pred_embed, *pred_inv_embed;
    const float *h0, *c0, *hp0, *cp0, *prelu_a;
    const float *W_sbert, *b_sbert, *W_cn, *b_cn;
    const float *W_gembed, *b_gembed, *W_goattn, *b_goattn, *W_fc, *b_fc;
    const float *lstm_Wih, *lstm_Whh, *lstm_bih, *lstm_bhh;
    const float *plstm_Wih, *plstm_Whh, *plstm_bih, *plstm_bhh;
    const float *W_act, *b_act, *W_obj1, *b_obj1, *W_obj2, *b_obj2;
    const float *W_state, *b_state, *W_act_inv, *b_act_inv, *W_obj1_inv, *b_obj1_inv;
};

// ---------------------------------------------------------------------------
// Kernel 2: entire post-reduction model head in one 512-thread block.
// NOTE: the reference's graph-attention softmax is over axis=1 of [N,1],
// which is identically 1.0 -> h_embed_raw is exactly the column sum of h.
// W_gattn / b_gattn are therefore (provably) dead inputs.
// ---------------------------------------------------------------------------
__global__ __launch_bounds__(512)
void head_kernel(HeadParams p, const float* __restrict__ partial, int nPartials,
                 float* __restrict__ out)
{
    __shared__ float sm[12032];                       // 48 KB pool, phase-reused
    const int tid  = threadIdx.x;
    const int wave = tid >> 5;
    const int lane = tid & 31;
    const float pa = p.prelu_a[0];

    float* HEMB = sm + 0;       // h_embed        [512] persistent
    float* GE   = sm + 512;     // goal_embed     [512] persistent
    float* GOB  = sm + 1024;    // goal_obj_embed [512] persistent
    float* HH   = sm + 1536;    // h_hist         [512] persistent
    float* C1   = sm + 2048;    // c1             [512] persistent
    float* PH   = sm + 2560;    // pred_hist      [512] persistent
    float* CP1  = sm + 3072;    // cp1            [512] persistent
    float* FIN  = sm + 3584;    // final          [512] persistent
    float* SCR  = sm + 4096;    // scratch        [6656]
    float* SB   = sm + 10752;   // small buffers  [1280]

    // ---- Phase A: reduce partial column sums (fixed order => deterministic)
    {
        float s = 0.f;
        for (int b = 0; b < nPartials; ++b) s += partial[(size_t)b * 512 + tid];
        SCR[tid] = s;                                   // sum_h at SCR[0..512)
    }
    __syncthreads();

    // ---- Phase B: goal_embed = prelu(W_sbert @ goalVec + b)   (K=384)
    {
        float* gv = SCR + 512;                          // [384]
        if (tid < 384) gv[tid] = p.goalVec[tid];
        __syncthreads();
        wmma_gemv(p.W_sbert, 512, 384, 384, 384, gv, p.b_sbert, GE, wave, lane, false);
        __syncthreads();
        GE[tid] = preluf(GE[tid], pa);
    }
    __syncthreads();

    // ---- Phase C: h_embed = prelu(W_gembed @ sum_h + b)       (K=512)
    wmma_gemv(p.W_gembed, 512, 512, 512, 512, SCR, p.b_gembed, HEMB, wave, lane, false);
    __syncthreads();
    HEMB[tid] = preluf(HEMB[tid], pa);
    __syncthreads();

    // ---- Phase D: go[8,512] = prelu(goalObjectsVec @ W_cn^T + b)  (K=300->320)
    float* gx = SCR;                                    // [8*320]  (sum_h dead)
    float* go = SCR + 2560;                             // [8*512]
    for (int i = tid; i < 8 * 320; i += 512) {
        int obj = i / 320, k = i - obj * 320;
        gx[i] = (k < 300) ? p.goalObjectsVec[obj * 300 + k] : 0.0f;
    }
    __syncthreads();
    for (int g = 0; g < 8; ++g)
        wmma_gemv(p.W_cn, 512, 300, 300, 320, gx + g * 320, p.b_cn, go + g * 512,
                  wave, lane, false);
    __syncthreads();
    for (int i = tid; i < 8 * 512; i += 512) go[i] = preluf(go[i], pa);
    __syncthreads();

    // ---- Phase E: goal-object attention (softmax over 8 objects)
    if (tid < 8) {
        float s = p.b_goattn[0];
        for (int j = 0; j < 512; ++j) s += p.W_goattn[j] * HEMB[j];
        const float* gg = go + tid * 512;
        for (int j = 0; j < 512; ++j) s += p.W_goattn[512 + j] * gg[j];
        SB[tid] = s;
    }
    __syncthreads();
    if (tid == 0) softmax_serial(SB, 8);
    __syncthreads();
    {
        float s = 0.f;
        for (int g = 0; g < 8; ++g) s += SB[g] * go[g * 512 + tid];
        GOB[tid] = s;
    }
    __syncthreads();

    // ---- Phase F/G: predicate-history LSTM (x = [pred, pred_inv], K=1222->1248)
    float* px = SCR;                                    // [1248]
    float* hx = SCR + 1248;                             // [512]
    float* z  = SCR + 1760;                             // [2048]
    for (int i = tid; i < 1248; i += 512)
        px[i] = (i < 611) ? p.pred_embed[i]
              : (i < 1222) ? p.pred_inv_embed[i - 611] : 0.0f;
    hx[tid] = p.hp0[tid];
    __syncthreads();
    wmma_gemv(p.plstm_Wih, 2048, 1222, 1222, 1248, px, p.plstm_bih, z, wave, lane, false);
    __syncthreads();
    wmma_gemv(p.plstm_Whh, 2048, 512, 512, 512, hx, p.plstm_bhh, z, wave, lane, true);
    __syncthreads();
    {
        float ig = sigmoidf(z[tid]);
        float fg = sigmoidf(z[512 + tid]);
        float gg = tanhf(z[1024 + tid]);
        float og = sigmoidf(z[1536 + tid]);
        float c2 = fg * p.cp0[tid] + ig * gg;
        PH[tid]  = og * tanhf(c2);
        CP1[tid] = c2;
    }
    __syncthreads();

    // ---- Phase H: graph-history LSTM (x = h_embed)
    hx[tid] = p.h0[tid];
    __syncthreads();
    wmma_gemv(p.lstm_Wih, 2048, 512, 512, 512, HEMB, p.lstm_bih, z, wave, lane, false);
    __syncthreads();
    wmma_gemv(p.lstm_Whh, 2048, 512, 512, 512, hx, p.lstm_bhh, z, wave, lane, true);
    __syncthreads();
    {
        float ig = sigmoidf(z[tid]);
        float fg = sigmoidf(z[512 + tid]);
        float gg = tanhf(z[1024 + tid]);
        float og = sigmoidf(z[1536 + tid]);
        float c2 = fg * p.c0[tid] + ig * gg;
        HH[tid] = og * tanhf(c2);
        C1[tid] = c2;
    }
    __syncthreads();

    // ---- Phase I: final = prelu(W_fc @ [h_embed,h_hist,gob,ge,ph] + b)  (K=2560)
    float* cat = SCR;                                   // [2560]
    for (int i = tid; i < 2560; i += 512) {
        float v;
        if      (i < 512)  v = HEMB[i];
        else if (i < 1024) v = HH[i - 512];
        else if (i < 1536) v = GOB[i - 1024];
        else if (i < 2048) v = GE[i - 1536];
        else               v = PH[i - 2048];
        cat[i] = v;
    }
    __syncthreads();
    wmma_gemv(p.W_fc, 512, 2560, 2560, 2560, cat, p.b_fc, FIN, wave, lane, false);
    __syncthreads();
    FIN[tid] = preluf(FIN[tid], pa);
    __syncthreads();

    // ---- Phase J: decode heads ------------------------------------------
    float* ACT = SB + 16;    // 11
    float* OHA = SB + 32;    // 11
    float* PB  = SB + 48;    // 128
    float* OHP = SB + 176;   // 128
    float* X2  = SB + 304;   // 651
    float* STB = SB + 960;   // 16

    for (int pass = 0; pass < 2; ++pass) {              // 0=forward, 1=inverse
        const float *Wa = pass ? p.W_act_inv : p.W_act;
        const float *ba = pass ? p.b_act_inv : p.b_act;
        const float *W1 = pass ? p.W_obj1_inv : p.W_obj1;
        const float *b1 = pass ? p.b_obj1_inv : p.b_obj1;
        const int base  = pass ? 283 : 0;
        const unsigned sA = pass ? 2u : 0u, sP = pass ? 3u : 1u;

        if (tid < 11) {
            float s = ba[tid];
            const float* w = Wa + tid * 512;
            for (int j = 0; j < 512; ++j) s += w[j] * FIN[j];
            ACT[tid] = s;
        }
        __syncthreads();
        if (tid == 0) softmax_serial(ACT, 11);
        __syncthreads();
        if (tid < 11) {
            out[base + tid] = ACT[tid];
            OHA[tid] = (ACT[tid] + gumbel_noise(tid, sA)) * TEMP_INV;
        }
        __syncthreads();
        if (tid == 0) softmax_serial(OHA, 11);
        __syncthreads();

        if (tid < 128) {                                 // obj1: K = 512+11
            const float* w = W1 + tid * 523;
            float s = b1[tid];
            for (int j = 0; j < 512; ++j) s += w[j] * FIN[j];
            for (int j = 0; j < 11;  ++j) s += w[512 + j] * OHA[j];
            PB[tid] = s;
        }
        __syncthreads();
        if (tid == 0) softmax_serial(PB, 128);
        __syncthreads();
        if (tid < 128) {
            out[base + 11 + tid] = PB[tid];
            OHP[tid] = (PB[tid] + gumbel_noise(tid, sP)) * TEMP_INV;
        }
        __syncthreads();
        if (tid == 0) softmax_serial(OHP, 128);
        __syncthreads();

        X2[tid] = FIN[tid];                              // build [final,oh_a,oh_p1]
        if (tid < 11)  X2[512 + tid] = OHA[tid];
        if (tid < 128) X2[523 + tid] = OHP[tid];
        __syncthreads();

        if (tid < 128) {                                 // obj2 (shared weights)
            const float* w = p.W_obj2 + tid * 651;
            float s = p.b_obj2[tid];
            for (int j = 0; j < 651; ++j) s += w[j] * X2[j];
            PB[tid] = s;
        }
        if (tid < 16) {                                  // state (shared weights)
            const float* w = p.W_state + tid * 651;
            float s = p.b_state[tid];
            for (int j = 0; j < 651; ++j) s += w[j] * X2[j];
            STB[tid] = s;
        }
        __syncthreads();
        if (tid == 0) { softmax_serial(PB, 128); softmax_serial(STB, 16); }
        __syncthreads();
        if (tid < 128) out[base + 139 + tid] = PB[tid];
        if (tid < 16)  out[base + 267 + tid] = STB[tid];
        __syncthreads();
    }

    // ---- LSTM state outputs: h_hist, c1, pred_hist, cp1
    out[566  + tid] = HH[tid];
    out[1078 + tid] = C1[tid];
    out[1590 + tid] = PH[tid];
    out[2102 + tid] = CP1[tid];
}

// ---------------------------------------------------------------------------
extern "C" void kernel_launch(void* const* d_in, const int* in_sizes, int n_in,
                              void* d_out, int out_size, void* d_ws, size_t ws_size,
                              hipStream_t stream)
{
    (void)n_in; (void)out_size;
    const float* h = (const float*)d_in[0];
    const int nRows = in_sizes[0] / 512;                // 100000

    // Partial column sums live in d_ws (nb * 512 floats); size grid from ws.
    size_t maxB = ws_size / (512 * sizeof(float));
    int nb = (maxB < 1) ? 1 : (maxB > 1024 ? 1024 : (int)maxB);
    int rpb = (nRows + nb - 1) / nb;
    float* partial = (float*)d_ws;

    colsum_kernel<<<nb, 256, 0, stream>>>(h, partial, nRows, rpb);

    HeadParams p;
    p.goalVec = (const float*)d_in[1];   p.goalObjectsVec = (const float*)d_in[2];
    p.pred_embed = (const float*)d_in[3]; p.pred_inv_embed = (const float*)d_in[4];
    p.h0 = (const float*)d_in[5];  p.c0 = (const float*)d_in[6];
    p.hp0 = (const float*)d_in[7]; p.cp0 = (const float*)d_in[8];
    p.prelu_a = (const float*)d_in[9];
    p.W_sbert = (const float*)d_in[10];  p.b_sbert = (const float*)d_in[11];
    p.W_cn = (const float*)d_in[12];     p.b_cn = (const float*)d_in[13];
    // d_in[14] W_gattn, d_in[15] b_gattn: provably dead (softmax over length-1 axis)
    p.W_gembed = (const float*)d_in[16]; p.b_gembed = (const float*)d_in[17];
    p.W_goattn = (const float*)d_in[18]; p.b_goattn = (const float*)d_in[19];
    p.W_fc = (const float*)d_in[20];     p.b_fc = (const float*)d_in[21];
    p.lstm_Wih = (const float*)d_in[22]; p.lstm_Whh = (const float*)d_in[23];
    p.lstm_bih = (const float*)d_in[24]; p.lstm_bhh = (const float*)d_in[25];
    p.plstm_Wih = (const float*)d_in[26]; p.plstm_Whh = (const float*)d_in[27];
    p.plstm_bih = (const float*)d_in[28]; p.plstm_bhh = (const float*)d_in[29];
    p.W_act = (const float*)d_in[30];    p.b_act = (const float*)d_in[31];
    p.W_obj1 = (const float*)d_in[32];   p.b_obj1 = (const float*)d_in[33];
    p.W_obj2 = (const float*)d_in[34];   p.b_obj2 = (const float*)d_in[35];
    p.W_state = (const float*)d_in[36];  p.b_state = (const float*)d_in[37];
    p.W_act_inv = (const float*)d_in[38]; p.b_act_inv = (const float*)d_in[39];
    p.W_obj1_inv = (const float*)d_in[40]; p.b_obj1_inv = (const float*)d_in[41];

    head_kernel<<<1, 512, 0, stream>>>(p, partial, nb, (float*)d_out);
}
